// FPN_LSS_Mamba_Mask_14370960572906
// MI455X (gfx1250) — matrix operand order repaired and checked by reading, compiled, not run on hardware
//
#include <hip/hip_runtime.h>
#include <hip/hip_bf16.h>
#include <math.h>

#define NSTATE 16

typedef __attribute__((ext_vector_type(16))) __bf16 bf16x16;
typedef __attribute__((ext_vector_type(8)))  __bf16 bf16x8;
typedef __attribute__((ext_vector_type(8)))  float  floatx8;
typedef __attribute__((ext_vector_type(4)))  int    v4i;

static inline unsigned cdivu(long long a, long long b) { return (unsigned)((a + b - 1) / b); }

#if __has_builtin(__builtin_amdgcn_global_load_async_to_lds_b128)
#define HAVE_ASYNC_LDS 1
#else
#define HAVE_ASYNC_LDS 0
#endif

static __device__ __forceinline__ void async_wait_all() {
#if HAVE_ASYNC_LDS
#if __has_builtin(__builtin_amdgcn_s_wait_asynccnt)
  __builtin_amdgcn_s_wait_asynccnt(0);
#else
  asm volatile("s_wait_asynccnt 0x0" ::: "memory");
#endif
#endif
}

// Unconditional 16-byte chunk global->LDS (async engine when available).
static __device__ __forceinline__ void stage_chunk_async(const float* gsrc, float* ldst) {
#if HAVE_ASYNC_LDS
  __builtin_amdgcn_global_load_async_to_lds_b128(
      (__attribute__((address_space(1))) v4i*)gsrc,
      (__attribute__((address_space(3))) v4i*)ldst, 0, 0);
#else
  *(float4*)ldst = *(const float4*)gsrc;
#endif
}

// ---------------------------------------------------------------------------
// Fragment build + WMMA from staged LDS tiles.
// A-frag (ISA 16-bit A 16x32): element e -> K = (e<8 ? 8g+e : 16+8g+(e-8))
// B-frag (ISA 16-bit B 32x16): element e -> K = (e<8 ? 8g+e : 16+8g+(e-8))
// ---------------------------------------------------------------------------
static __device__ __forceinline__ floatx8
tile_mma(const float (*As)[36], const __bf16 (*Bs)[40], int w, int lane, floatx8 acc) {
  int g = lane >> 4, r16 = lane & 15;
  const float* ap = &As[w * 16 + r16][0];
  float4 a0 = *(const float4*)(ap + 8 * g);
  float4 a1 = *(const float4*)(ap + 8 * g + 4);
  float4 a2 = *(const float4*)(ap + 16 + 8 * g);
  float4 a3 = *(const float4*)(ap + 16 + 8 * g + 4);
  bf16x16 af;
  af[0] = (__bf16)a0.x;  af[1] = (__bf16)a0.y;  af[2] = (__bf16)a0.z;  af[3] = (__bf16)a0.w;
  af[4] = (__bf16)a1.x;  af[5] = (__bf16)a1.y;  af[6] = (__bf16)a1.z;  af[7] = (__bf16)a1.w;
  af[8] = (__bf16)a2.x;  af[9] = (__bf16)a2.y;  af[10] = (__bf16)a2.z; af[11] = (__bf16)a2.w;
  af[12] = (__bf16)a3.x; af[13] = (__bf16)a3.y; af[14] = (__bf16)a3.z; af[15] = (__bf16)a3.w;
  const __bf16* bp = &Bs[r16][0];
  bf16x8 b0 = *(const bf16x8*)(bp + 8 * g);
  bf16x8 b1 = *(const bf16x8*)(bp + 16 + 8 * g);
  bf16x16 bv;
#pragma unroll
  for (int e = 0; e < 8; ++e) { bv[e] = b0[e]; bv[e + 8] = b1[e]; }
  return __builtin_amdgcn_wmma_f32_16x16x32_bf16(false, af, false, bv, (short)0, acc,
                                                 false, false);
}

// ---------------------------------------------------------------------------
// GEMM: Out[M,N] = A[M,K](fp32, ld=lda) @ W[K,N](bf16, ld=N) + bias.
// 128 threads = 4 waves; block tile 64x16; K staged in steps of 32 via LDS.
// REQUIRES: M % 64 == 0, K % 32 == 0, lda % 4 == 0, row length of A >= K.
// ep: 0=none, 1=softplus.
// ---------------------------------------------------------------------------
__global__ __launch_bounds__(128)
void k_gemm_wmma(const float* __restrict__ A, const __bf16* __restrict__ W,
                 const float* __restrict__ bias, float* __restrict__ Out,
                 int M, int N, int K, int lda, int ldc, int ep) {
  __shared__ __align__(16) float  As[64][36];
  __shared__ __align__(16) __bf16 Bs[16][40];
  int tid = threadIdx.x, lane = tid & 31, w = tid >> 5;
  int g = lane >> 4, r16 = lane & 15;
  int tm = blockIdx.y * 64, tn = blockIdx.x * 16;
  // A staging role: row = tid/2 (0..63), half = tid&1 covers 16 of 32 K-cols
  int ar = tid >> 1, ahalf = tid & 1;
  const float* arow = A + (size_t)(tm + ar) * lda + ahalf * 16;
  float* lrow = &As[ar][ahalf * 16];
  floatx8 acc = {};
  for (int k0 = 0; k0 < K; k0 += 32) {
#pragma unroll
    for (int c = 0; c < 4; ++c) stage_chunk_async(arow + k0 + c * 4, lrow + c * 4);
#pragma unroll
    for (int i = 0; i < 4; ++i) {
      int e2 = tid * 4 + i;
      int kk = e2 >> 4, nn = e2 & 15;
      int ng = tn + nn;
      int ngc = ng < N ? ng : N - 1;                     // clamped, branchless
      __bf16 v = W[(size_t)(k0 + kk) * N + ngc];
      Bs[nn][kk] = (ng < N) ? v : (__bf16)0.f;
    }
    async_wait_all();
    __syncthreads();
    acc = tile_mma(As, Bs, w, lane, acc);
    __syncthreads();
  }
  int n = tn + r16;
  if (n < N) {
    float bb = bias ? bias[n] : 0.f;
#pragma unroll
    for (int r = 0; r < 8; ++r) {
      int mm = tm + w * 16 + r + 8 * g;  // C/D: VGPR r -> M = r + 8*(lane>>4)
      float v = acc[r] + bb;
      if (ep == 1) v = (v > 20.f) ? v : log1pf(__expf(v));  // softplus
      Out[(size_t)mm * ldc + n] = v;
    }
  }
}

// ---------------------------------------------------------------------------
// 3x3 SAME conv, implicit GEMM: In NHWC fp32, Wr bf16 [9][Cin][Cout], Out NHWC.
// Accumulator carried across 9 taps x K-steps.
// REQUIRES: (Bn*H*Wd) % 64 == 0, Cin % 32 == 0, Cout % 16 == 0.
// ---------------------------------------------------------------------------
__global__ __launch_bounds__(128)
void k_conv3x3_wmma(const float* __restrict__ In, const __bf16* __restrict__ Wr,
                    float* __restrict__ Out, int Bn, int H, int Wd,
                    int Cin, int Cout) {
  __shared__ __align__(16) float  As[64][36];
  __shared__ __align__(16) __bf16 Bs[16][40];
  int tid = threadIdx.x, lane = tid & 31, w = tid >> 5;
  int g = lane >> 4, r16 = lane & 15;
  int tm = blockIdx.y * 64, tn = blockIdx.x * 16;
  int M = Bn * H * Wd;
  int ar = tid >> 1, ahalf = tid & 1;
  int am = tm + ar;
  int b = am / (H * Wd);
  int p = am - b * (H * Wd);
  int yy = p / Wd;
  int xx = p - yy * Wd;
  float* lrow = &As[ar][ahalf * 16];
  floatx8 acc = {};
  for (int t = 0; t < 9; ++t) {
    int dy = t / 3 - 1, dx = t % 3 - 1;
    int ys = yy + dy, xs = xx + dx;
    bool inb = ys >= 0 && ys < H && xs >= 0 && xs < Wd;
    const float* arow =
        In + (((size_t)b * H + (inb ? ys : 0)) * Wd + (inb ? xs : 0)) * Cin + ahalf * 16;
    const __bf16* wt = Wr + (size_t)t * Cin * Cout;
    for (int k0 = 0; k0 < Cin; k0 += 32) {
      if (inb) {  // one divergence point per row per K-step
#pragma unroll
        for (int c = 0; c < 4; ++c) stage_chunk_async(arow + k0 + c * 4, lrow + c * 4);
      } else {
#pragma unroll
        for (int c = 0; c < 4; ++c) *(float4*)(lrow + c * 4) = make_float4(0.f, 0.f, 0.f, 0.f);
      }
#pragma unroll
      for (int i = 0; i < 4; ++i) {
        int e2 = tid * 4 + i;
        int kk = e2 >> 4, nn = e2 & 15;
        Bs[nn][kk] = wt[(size_t)(k0 + kk) * Cout + tn + nn];
      }
      async_wait_all();
      __syncthreads();
      acc = tile_mma(As, Bs, w, lane, acc);
      __syncthreads();
    }
  }
  int n = tn + r16;
#pragma unroll
  for (int r = 0; r < 8; ++r) {
    int mm = tm + w * 16 + r + 8 * g;
    Out[(size_t)mm * Cout + n] = acc[r];
  }
}

// ---------------------------------------------------------------------------
// LayerNorm over last dim, one wave per row (wave32 shuffle reduction).
// ---------------------------------------------------------------------------
__global__ __launch_bounds__(32)
void k_layernorm(const float* __restrict__ x, const float* __restrict__ gg,
                 const float* __restrict__ bb, float* __restrict__ y,
                 int rows, int C) {
  int row = blockIdx.x;
  if (row >= rows) return;
  int lane = threadIdx.x;
  const float* xr = x + (size_t)row * C;
  float s = 0.f;
  for (int c = lane; c < C; c += 32) s += xr[c];
  for (int o = 16; o > 0; o >>= 1) s += __shfl_xor(s, o, 32);
  float mu = s / (float)C;
  float v = 0.f;
  for (int c = lane; c < C; c += 32) { float d = xr[c] - mu; v += d * d; }
  for (int o = 16; o > 0; o >>= 1) v += __shfl_xor(v, o, 32);
  float inv = rsqrtf(v / (float)C + 1e-5f);
  float* yr = y + (size_t)row * C;
  for (int c = lane; c < C; c += 32) yr[c] = (xr[c] - mu) * inv * gg[c] + bb[c];
}

// Depthwise 3x3 SAME conv + bias + SiLU, NHWC.
__global__ void k_dwconv_silu(const float* __restrict__ In,
                              const float* __restrict__ Kw,
                              const float* __restrict__ bias,
                              float* __restrict__ Out,
                              int Bn, int H, int Wd, int C) {
  size_t i = blockIdx.x * (size_t)blockDim.x + threadIdx.x;
  size_t total = (size_t)Bn * H * Wd * C;
  if (i >= total) return;
  int c = (int)(i % C);
  size_t p = i / C;
  int x = (int)(p % Wd); p /= Wd;
  int y = (int)(p % H);
  int b = (int)(p / H);
  float s = bias[c];
  for (int dy = -1; dy <= 1; ++dy)
    for (int dx = -1; dx <= 1; ++dx) {
      int ys = y + dy, xs = x + dx;
      if (ys < 0 || ys >= H || xs < 0 || xs >= Wd) continue;
      s += In[(((size_t)b * H + ys) * Wd + xs) * C + c] * Kw[c * 9 + (dy + 1) * 3 + (dx + 1)];
    }
  Out[i] = s / (1.f + __expf(-s));  // silu
}

// Selective scan, one thread per (batch, channel); 16-state SSM in registers.
__global__ void k_scan(const float* __restrict__ u, const float* __restrict__ dtp,
                       const float* __restrict__ A_log, const float* __restrict__ xdbl,
                       int ldx, int R, const float* __restrict__ D,
                       float* __restrict__ out, int Bn, int L, int C,
                       int rev_out, int accum) {
  int idx = blockIdx.x * blockDim.x + threadIdx.x;
  if (idx >= Bn * C) return;
  int b = idx / C, c = idx % C;
  float h[NSTATE], An[NSTATE];
#pragma unroll
  for (int n = 0; n < NSTATE; ++n) { An[n] = -__expf(A_log[c * NSTATE + n]); h[n] = 0.f; }
  float Dc = D[c];
  for (int l = 0; l < L; ++l) {
    size_t row = (size_t)b * L + l;
    float dt = dtp[row * C + c];
    float uu = u[row * C + c];
    const float* xr = xdbl + row * ldx + R;
    float du = dt * uu, y = 0.f;
#pragma unroll
    for (int n = 0; n < NSTATE; ++n) {
      h[n] = __expf(dt * An[n]) * h[n] + du * xr[n];
      y += h[n] * xr[NSTATE + n];
    }
    y += uu * Dc;
    size_t orow = (size_t)b * L + (rev_out ? (L - 1 - l) : l);
    float* o = out + orow * C + c;
    if (accum) *o += y; else *o = y;
  }
}

// Elementwise helpers --------------------------------------------------------
__global__ void k_add(float* __restrict__ dst, const float* __restrict__ src, size_t n) {
  size_t i = blockIdx.x * (size_t)blockDim.x + threadIdx.x;
  if (i < n) dst[i] += src[i];
}

__global__ void k_reverse(const float* __restrict__ src, float* __restrict__ dst,
                          int Bn, int L, int C) {
  size_t i = blockIdx.x * (size_t)blockDim.x + threadIdx.x;
  size_t total = (size_t)Bn * L * C;
  if (i >= total) return;
  int c = (int)(i % C);
  size_t r = i / C;
  int l = (int)(r % L);
  int b = (int)(r / L);
  dst[((size_t)b * L + (L - 1 - l)) * C + c] = src[i];
}

// out[b][w][h][c] = in[b][h][w][c]; in dims (B,Hin,Win,C)
__global__ void k_transpose_hw(const float* __restrict__ in, float* __restrict__ out,
                               int Bn, int Hin, int Win, int C) {
  size_t i = blockIdx.x * (size_t)blockDim.x + threadIdx.x;
  size_t total = (size_t)Bn * Hin * Win * C;
  if (i >= total) return;
  int c = (int)(i % C);
  size_t r = i / C;
  int h = (int)(r % Hin); r /= Hin;
  int w = (int)(r % Win);
  int b = (int)(r / Win);
  out[i] = in[(((size_t)b * Hin + h) * Win + w) * C + c];
}

__global__ void k_nchw_to_nhwc(const float* __restrict__ in, float* __restrict__ out,
                               int Bn, int C, int H, int Wd) {
  size_t i = blockIdx.x * (size_t)blockDim.x + threadIdx.x;
  size_t total = (size_t)Bn * C * H * Wd;
  if (i >= total) return;
  int c = (int)(i % C);
  size_t r = i / C;
  int x = (int)(r % Wd); r /= Wd;
  int y = (int)(r % H);
  int b = (int)(r / H);
  out[i] = in[(((size_t)b * C + c) * H + y) * Wd + x];
}

__global__ void k_nhwc_to_nchw(const float* __restrict__ in, float* __restrict__ out,
                               int Bn, int H, int Wd, int C) {
  size_t i = blockIdx.x * (size_t)blockDim.x + threadIdx.x;
  size_t total = (size_t)Bn * C * H * Wd;
  if (i >= total) return;
  int x = (int)(i % Wd);
  size_t r = i / Wd;
  int y = (int)(r % H); r /= H;
  int c = (int)(r % C);
  int b = (int)(r / C);
  out[i] = in[(((size_t)b * H + y) * Wd + x) * C + c];
}

// Bilinear upsample (align-corners formula of reference). nhwc flag picks layout.
__global__ void k_upsample(const float* __restrict__ in, float* __restrict__ out,
                           int Bn, int C, int H, int Wd, int s, int nhwc) {
  int Ho = H * s, Wo = Wd * s;
  size_t total = (size_t)Bn * C * Ho * Wo;
  size_t i = blockIdx.x * (size_t)blockDim.x + threadIdx.x;
  if (i >= total) return;
  int b, c, yo, xo;
  if (nhwc) {
    c = (int)(i % C); size_t r = i / C;
    xo = (int)(r % Wo); r /= Wo;
    yo = (int)(r % Ho); b = (int)(r / Ho);
  } else {
    xo = (int)(i % Wo); size_t r = i / Wo;
    yo = (int)(r % Ho); r /= Ho;
    c = (int)(r % C); b = (int)(r / C);
  }
  float fy = (Ho > 1) ? yo * (float)(H - 1) / (float)(Ho - 1) : 0.f;
  float fx = (Wo > 1) ? xo * (float)(Wd - 1) / (float)(Wo - 1) : 0.f;
  int y0 = (int)floorf(fy), x0 = (int)floorf(fx);
  int y1 = min(y0 + 1, H - 1), x1 = min(x0 + 1, Wd - 1);
  float wy = fy - (float)y0, wx = fx - (float)x0;
  auto ld = [&](int yy, int xx) -> float {
    return nhwc ? in[(((size_t)b * H + yy) * Wd + xx) * C + c]
                : in[(((size_t)b * C + c) * H + yy) * Wd + xx];
  };
  out[i] = ld(y0, x0) * (1 - wy) * (1 - wx) + ld(y0, x1) * (1 - wy) * wx +
           ld(y1, x0) * wy * (1 - wx) + ld(y1, x1) * wy * wx;
}

// concat([NCHW Ca], [NCHW Cb]) -> NHWC (Ca+Cb)
__global__ void k_concat(const float* __restrict__ a, const float* __restrict__ b2,
                         float* __restrict__ out, int Bn, int H, int Wd, int Ca, int Cb) {
  int C = Ca + Cb;
  size_t total = (size_t)Bn * H * Wd * C;
  size_t i = blockIdx.x * (size_t)blockDim.x + threadIdx.x;
  if (i >= total) return;
  int c = (int)(i % C);
  size_t r = i / C;
  int x = (int)(r % Wd); r /= Wd;
  int y = (int)(r % H);
  int b = (int)(r / H);
  out[i] = (c < Ca) ? a[(((size_t)b * Ca + c) * H + y) * Wd + x]
                    : b2[(((size_t)b * Cb + (c - Ca)) * H + y) * Wd + x];
}

// fp32 -> bf16 (flat copy)
__global__ void k_cvt_bf16(const float* __restrict__ in, __bf16* __restrict__ out, size_t n) {
  size_t i = blockIdx.x * (size_t)blockDim.x + threadIdx.x;
  if (i < n) out[i] = (__bf16)in[i];
}

// fp32 [N][K] -> bf16 [K][N]  (for f_w OI -> [K=I][N=O])
__global__ void k_cvt_bf16_t(const float* __restrict__ in, __bf16* __restrict__ out,
                             int N, int K) {
  size_t i = blockIdx.x * (size_t)blockDim.x + threadIdx.x;
  size_t total = (size_t)N * K;
  if (i >= total) return;
  int n = (int)(i % N);
  int k = (int)(i / N);
  out[i] = (__bf16)in[(size_t)n * K + k];
}

// dt_w fp32 [2][R][C] -> bf16 [2][32][C], zero-padded rows R..31
__global__ void k_cvt_pad_dt(const float* __restrict__ in, __bf16* __restrict__ out,
                             int R, int C) {
  size_t i = blockIdx.x * (size_t)blockDim.x + threadIdx.x;
  size_t total = (size_t)2 * 32 * C;
  if (i >= total) return;
  int c = (int)(i % C);
  size_t r = i / C;
  int k = (int)(r % 32);
  int d = (int)(r / 32);
  out[i] = (k < R) ? (__bf16)in[((size_t)d * R + k) * C + c] : (__bf16)0.f;
}

// OIHW (Cout,Cin,3,3) fp32 -> bf16 [9][Cin][Cout]
__global__ void k_repack_oihw(const float* __restrict__ w, __bf16* __restrict__ out,
                              int Cout, int Cin) {
  size_t total = (size_t)9 * Cin * Cout;
  size_t i = blockIdx.x * (size_t)blockDim.x + threadIdx.x;
  if (i >= total) return;
  int co = (int)(i % Cout);
  size_t r = i / Cout;
  int ci = (int)(r % Cin);
  int t = (int)(r / Cin);
  out[i] = (__bf16)w[((size_t)co * Cin + ci) * 9 + t];
}

// BatchNorm statistics (mean over B*H*W per channel); one block per channel.
__global__ __launch_bounds__(256)
void k_bn_stats(const float* __restrict__ In, float* __restrict__ mean,
                float* __restrict__ inv, int rows, int C) {
  int c = blockIdx.x;
  __shared__ float ss[256], sq[256];
  float s = 0.f, q = 0.f;
  for (int r = threadIdx.x; r < rows; r += 256) {
    float v = In[(size_t)r * C + c];
    s += v; q += v * v;
  }
  ss[threadIdx.x] = s; sq[threadIdx.x] = q;
  __syncthreads();
  for (int o = 128; o > 0; o >>= 1) {
    if ((int)threadIdx.x < o) { ss[threadIdx.x] += ss[threadIdx.x + o]; sq[threadIdx.x] += sq[threadIdx.x + o]; }
    __syncthreads();
  }
  if (threadIdx.x == 0) {
    float mu = ss[0] / (float)rows;
    float var = sq[0] / (float)rows - mu * mu;
    mean[c] = mu;
    inv[c] = rsqrtf(var + 1e-5f);
  }
}

__global__ void k_bn_relu(float* __restrict__ x, const float* __restrict__ mean,
                          const float* __restrict__ inv, const float* __restrict__ g,
                          const float* __restrict__ b, size_t total, int C) {
  size_t i = blockIdx.x * (size_t)blockDim.x + threadIdx.x;
  if (i >= total) return;
  int c = (int)(i % C);
  float v = (x[i] - mean[c]) * inv[c] * g[c] + b[c];
  x[i] = v > 0.f ? v : 0.f;
}

// ---------------------------------------------------------------------------
// Host-side orchestration
// ---------------------------------------------------------------------------
struct VSSP {
  const float *A_log, *D, *dt_b, *dt_w, *dw_b, *dw_k, *in_b, *in_w,
              *ln_b, *ln_g, *out_b, *out_ln_b, *out_ln_g, *out_w, *x_proj_w;
  int C, R;
};

static VSSP make_vssp(void* const* d_in, int base, int C) {
  VSSP p;
  p.A_log    = (const float*)d_in[base + 0];
  p.D        = (const float*)d_in[base + 1];
  p.dt_b     = (const float*)d_in[base + 2];
  p.dt_w     = (const float*)d_in[base + 3];
  p.dw_b     = (const float*)d_in[base + 4];
  p.dw_k     = (const float*)d_in[base + 5];
  p.in_b     = (const float*)d_in[base + 6];
  p.in_w     = (const float*)d_in[base + 7];
  p.ln_b     = (const float*)d_in[base + 8];
  p.ln_g     = (const float*)d_in[base + 9];
  p.out_b    = (const float*)d_in[base + 10];
  p.out_ln_b = (const float*)d_in[base + 11];
  p.out_ln_g = (const float*)d_in[base + 12];
  p.out_w    = (const float*)d_in[base + 13];
  p.x_proj_w = (const float*)d_in[base + 14];
  p.C = C;
  p.R = (C / 16 > 1) ? C / 16 : 1;
  return p;
}

struct Scratch {
  float *t1, *t2, *t3, *t4, *t5, *t6, *xdbl, *dtp;
  __bf16 *wb_in, *wb_xp, *wb_dt, *wb_out;
};

static void run_vss(hipStream_t stream, float* xbuf, int Bn, int H, int Wd,
                    const VSSP& p, int col_scan, const Scratch& S) {
  int C = p.C, R = p.R;
  int L = H * Wd, rows = Bn * L, ldx = R + 2 * NSTATE;
  size_t tot = (size_t)rows * C;
  dim3 gC(cdivu(C, 16), cdivu(rows, 64));

  // pre-convert this block's weights to bf16 (tiny); dt_w zero-padded to K=32
  k_cvt_bf16<<<cdivu((size_t)C * C, 256), 256, 0, stream>>>(p.in_w, S.wb_in, (size_t)C * C);
  k_cvt_bf16<<<cdivu((size_t)2 * C * ldx, 256), 256, 0, stream>>>(p.x_proj_w, S.wb_xp, (size_t)2 * C * ldx);
  k_cvt_pad_dt<<<cdivu((size_t)2 * 32 * C, 256), 256, 0, stream>>>(p.dt_w, S.wb_dt, R, C);
  k_cvt_bf16<<<cdivu((size_t)C * C, 256), 256, 0, stream>>>(p.out_w, S.wb_out, (size_t)C * C);

  k_layernorm<<<rows, 32, 0, stream>>>(xbuf, p.ln_g, p.ln_b, S.t1, rows, C);
  k_gemm_wmma<<<gC, 128, 0, stream>>>(S.t1, S.wb_in, p.in_b, S.t2, rows, C, C, C, C, 0);
  k_dwconv_silu<<<cdivu(tot, 256), 256, 0, stream>>>(S.t2, p.dw_k, p.dw_b, S.t3, Bn, H, Wd, C);

  float* seq = S.t3;
  if (col_scan) {
    k_transpose_hw<<<cdivu(tot, 256), 256, 0, stream>>>(S.t3, S.t4, Bn, H, Wd, C);
    seq = S.t4;
  }

  for (int k = 0; k < 2; ++k) {
    const float* sp = seq;
    if (k == 1) {
      k_reverse<<<cdivu(tot, 256), 256, 0, stream>>>(seq, S.t5, Bn, L, C);
      sp = S.t5;
    }
    k_gemm_wmma<<<dim3(cdivu(ldx, 16), cdivu(rows, 64)), 128, 0, stream>>>(
        sp, S.wb_xp + (size_t)k * C * ldx, nullptr, S.xdbl, rows, ldx, C, C, ldx, 0);
    // dt projection: K padded to 32 (weight rows R..31 are zero); A rows are
    // xdbl rows (length ldx >= 32), so reading K=32 columns stays in-bounds.
    k_gemm_wmma<<<gC, 128, 0, stream>>>(
        S.xdbl, S.wb_dt + (size_t)k * 32 * C, p.dt_b + (size_t)k * C, S.dtp,
        rows, C, 32, ldx, C, /*softplus*/1);
    k_scan<<<cdivu(Bn * C, 128), 128, 0, stream>>>(
        sp, S.dtp, p.A_log + (size_t)k * C * NSTATE, S.xdbl, ldx, R,
        p.D + (size_t)k * C, S.t6, Bn, L, C, /*rev_out*/k, /*accum*/k);
  }

  float* yb = S.t6;
  if (col_scan) {
    k_transpose_hw<<<cdivu(tot, 256), 256, 0, stream>>>(S.t6, S.t3, Bn, Wd, H, C);
    yb = S.t3;
  }

  k_layernorm<<<rows, 32, 0, stream>>>(yb, p.out_ln_g, p.out_ln_b, S.t1, rows, C);
  k_gemm_wmma<<<gC, 128, 0, stream>>>(S.t1, S.wb_out, p.out_b, S.t2, rows, C, C, C, C, 0);
  k_add<<<cdivu(tot, 256), 256, 0, stream>>>(xbuf, S.t2, tot);
}

extern "C" void kernel_launch(void* const* d_in, const int* in_sizes, int n_in,
                              void* d_out, int out_size, void* d_ws, size_t ws_size,
                              hipStream_t stream) {
  (void)in_sizes; (void)n_in; (void)out_size; (void)ws_size;
  const int Bn = 2;
  const int C1 = 256, C0 = 64, CIN = 320, CMID = 512, CO = 256;

  const float* feat0 = (const float*)d_in[0];   // (2,64,64,64) NCHW
  const float* feat2 = (const float*)d_in[2];   // (2,256,16,16) NCHW
  VSSP b1ccw = make_vssp(d_in, 3 + 0 * 15, C1);
  VSSP b1cw  = make_vssp(d_in, 3 + 1 * 15, C1);
  VSSP b2ccw = make_vssp(d_in, 3 + 2 * 15, C1);
  VSSP b2cw  = make_vssp(d_in, 3 + 3 * 15, C1);
  VSSP b3ccw = make_vssp(d_in, 3 + 4 * 15, CIN);
  VSSP b3cw  = make_vssp(d_in, 3 + 5 * 15, CIN);
  VSSP b4ccw = make_vssp(d_in, 3 + 6 * 15, CIN);
  VSSP b4cw  = make_vssp(d_in, 3 + 7 * 15, CIN);
  const float* bn1_b = (const float*)d_in[123];
  const float* bn1_g = (const float*)d_in[124];
  const float* bn2_b = (const float*)d_in[125];
  const float* bn2_g = (const float*)d_in[126];
  const float* bn3_b = (const float*)d_in[127];
  const float* bn3_g = (const float*)d_in[128];
  const float* c1_w  = (const float*)d_in[129];
  const float* c2_w  = (const float*)d_in[130];
  const float* f_b   = (const float*)d_in[131];
  const float* f_w   = (const float*)d_in[132];
  const float* u_w   = (const float*)d_in[133];

  float* dout = (float*)d_out;

  // -------- workspace bump allocation (bytes, 256B aligned) --------
  char* wsb = (char*)d_ws;
  size_t off = 0;
  auto alloc = [&](size_t bytes) {
    void* p = wsb + off;
    off += (bytes + 255) & ~(size_t)255;
    return p;
  };
  const size_t R2 = (size_t)Bn * 64 * 64 * CIN;  // 2,621,440 elems
  float* xbuf1 = (float*)alloc((size_t)Bn * 16 * 16 * C1 * 4);
  float* xbuf2 = (float*)alloc(R2 * 4);
  Scratch S;
  S.t1 = (float*)alloc(R2 * 4); S.t2 = (float*)alloc(R2 * 4);
  S.t3 = (float*)alloc(R2 * 4); S.t4 = (float*)alloc(R2 * 4);
  S.t5 = (float*)alloc(R2 * 4); S.t6 = (float*)alloc(R2 * 4);
  S.xdbl = (float*)alloc((size_t)Bn * 64 * 64 * 64 * 4);
  S.dtp  = (float*)alloc(R2 * 4);
  S.wb_in  = (__bf16*)alloc((size_t)CIN * CIN * 2);
  S.wb_xp  = (__bf16*)alloc((size_t)2 * CIN * 64 * 2);
  S.wb_dt  = (__bf16*)alloc((size_t)2 * 32 * CIN * 2);
  S.wb_out = (__bf16*)alloc((size_t)CIN * CIN * 2);
  float* x1n   = (float*)alloc((size_t)Bn * C1 * 16 * 16 * 4);
  float* x1up  = (float*)alloc((size_t)Bn * C1 * 64 * 64 * 4);
  float* cmid1 = (float*)alloc((size_t)Bn * 64 * 64 * CMID * 4);
  float* cmid2 = (float*)alloc((size_t)Bn * 64 * 64 * CMID * 4);
  float* cup   = (float*)alloc((size_t)Bn * 128 * 128 * CMID * 4);
  float* cu    = (float*)alloc((size_t)Bn * 128 * 128 * CO * 4);
  float* cf    = (float*)alloc((size_t)Bn * 128 * 128 * CO * 4);
  __bf16* wrp  = (__bf16*)alloc((size_t)9 * CMID * CMID * 2);
  __bf16* wf   = (__bf16*)alloc((size_t)CO * CO * 2);
  float* bns   = (float*)alloc(1024 * 4);

  // -------- stage 1: feat2 -> NHWC, 4 VSS blocks @ C=256, 16x16 --------
  size_t n1 = (size_t)Bn * C1 * 16 * 16;
  k_nchw_to_nhwc<<<cdivu(n1, 256), 256, 0, stream>>>(feat2, xbuf1, Bn, C1, 16, 16);
  run_vss(stream, xbuf1, Bn, 16, 16, b1ccw, 0, S);
  run_vss(stream, xbuf1, Bn, 16, 16, b1cw,  1, S);
  run_vss(stream, xbuf1, Bn, 16, 16, b2ccw, 0, S);
  run_vss(stream, xbuf1, Bn, 16, 16, b2cw,  1, S);

  // bev_mask_feat (NCHW) -> d_out tail; upsample x4; concat with feat0 (NHWC)
  k_nhwc_to_nchw<<<cdivu(n1, 256), 256, 0, stream>>>(xbuf1, x1n, Bn, 16, 16, C1);
  (void)hipMemcpyAsync(dout + (size_t)Bn * CO * 128 * 128, x1n, n1 * sizeof(float),
                       hipMemcpyDeviceToDevice, stream);
  size_t nup1 = (size_t)Bn * C1 * 64 * 64;
  k_upsample<<<cdivu(nup1, 256), 256, 0, stream>>>(x1n, x1up, Bn, C1, 16, 16, 4, 0);
  k_concat<<<cdivu(R2, 256), 256, 0, stream>>>(feat0, x1up, xbuf2, Bn, 64, 64, C0, C1);

  // -------- stage 2: 4 VSS blocks @ C=320, 64x64 --------
  run_vss(stream, xbuf2, Bn, 64, 64, b3ccw, 0, S);
  run_vss(stream, xbuf2, Bn, 64, 64, b3cw,  1, S);
  run_vss(stream, xbuf2, Bn, 64, 64, b4ccw, 0, S);
  run_vss(stream, xbuf2, Bn, 64, 64, b4cw,  1, S);

  // -------- conv head (NHWC, implicit-GEMM WMMA convs) --------
  int rows64 = Bn * 64 * 64;     // 8192
  int rows128 = Bn * 128 * 128;  // 32768
  // c1: 320 -> 512, bn1, relu
  k_repack_oihw<<<cdivu((size_t)9 * CIN * CMID, 256), 256, 0, stream>>>(c1_w, wrp, CMID, CIN);
  k_conv3x3_wmma<<<dim3(cdivu(CMID, 16), cdivu(rows64, 64)), 128, 0, stream>>>(
      xbuf2, wrp, cmid1, Bn, 64, 64, CIN, CMID);
  k_bn_stats<<<CMID, 256, 0, stream>>>(cmid1, bns, bns + 512, rows64, CMID);
  k_bn_relu<<<cdivu((size_t)rows64 * CMID, 256), 256, 0, stream>>>(
      cmid1, bns, bns + 512, bn1_g, bn1_b, (size_t)rows64 * CMID, CMID);
  // c2: 512 -> 512, bn2, relu
  k_repack_oihw<<<cdivu((size_t)9 * CMID * CMID, 256), 256, 0, stream>>>(c2_w, wrp, CMID, CMID);
  k_conv3x3_wmma<<<dim3(cdivu(CMID, 16), cdivu(rows64, 64)), 128, 0, stream>>>(
      cmid1, wrp, cmid2, Bn, 64, 64, CMID, CMID);
  k_bn_stats<<<CMID, 256, 0, stream>>>(cmid2, bns, bns + 512, rows64, CMID);
  k_bn_relu<<<cdivu((size_t)rows64 * CMID, 256), 256, 0, stream>>>(
      cmid2, bns, bns + 512, bn2_g, bn2_b, (size_t)rows64 * CMID, CMID);
  // upsample x2 (NHWC)
  size_t nup2 = (size_t)rows128 * CMID;
  k_upsample<<<cdivu(nup2, 256), 256, 0, stream>>>(cmid2, cup, Bn, CMID, 64, 64, 2, 1);
  // u: 512 -> 256, bn3, relu
  k_repack_oihw<<<cdivu((size_t)9 * CMID * CO, 256), 256, 0, stream>>>(u_w, wrp, CO, CMID);
  k_conv3x3_wmma<<<dim3(cdivu(CO, 16), cdivu(rows128, 64)), 128, 0, stream>>>(
      cup, wrp, cu, Bn, 128, 128, CMID, CO);
  k_bn_stats<<<CO, 256, 0, stream>>>(cu, bns, bns + 512, rows128, CO);
  k_bn_relu<<<cdivu((size_t)rows128 * CO, 256), 256, 0, stream>>>(
      cu, bns, bns + 512, bn3_g, bn3_b, (size_t)rows128 * CO, CO);
  // f: 1x1 conv == GEMM; f_w OIHW(256,256,1,1) -> bf16 [K=in][N=out]
  k_cvt_bf16_t<<<cdivu((size_t)CO * CO, 256), 256, 0, stream>>>(f_w, wf, CO, CO);
  k_gemm_wmma<<<dim3(cdivu(CO, 16), cdivu(rows128, 64)), 128, 0, stream>>>(
      cu, wf, f_b, cf, rows128, CO, CO, CO, CO, 0);
  // NHWC -> NCHW into d_out (output 0)
  k_nhwc_to_nchw<<<cdivu((size_t)rows128 * CO, 256), 256, 0, stream>>>(
      cf, dout, Bn, 128, 128, CO);
}
